// NonLocalAttentionStack_27839978013240
// MI455X (gfx1250) — matrix operand order repaired and checked by reading, not compiled, not run on hardware
//
#include <hip/hip_runtime.h>
#include <hip/hip_bf16.h>

// ---------------------------------------------------------------------------
// NonLocalAttentionStack on gfx1250 (MI455X), wave32 + WMMA f16 pipeline.
//   K1: LayerNorm -> x_ln (36864 x 128) f16, pixel-major
//   K2: fused QKV GEMM (M=36864, N=384, K=128) via v_wmma_f32_16x16x32_f16
//   K3: non-local search as unfold-GEMM: per query row, WMMA GEMM
//       (M=16 queries x N=160 candidate band x K=288 patch features),
//       then register-resident top-16 + softmax. Box-sum folded into K dim.
//   K4: gather+weight+grouped-Conv3d as implicit WMMA GEMM (K=4608 per group)
// ---------------------------------------------------------------------------

typedef __attribute__((ext_vector_type(16))) _Float16 v16h;
typedef __attribute__((ext_vector_type(8)))  _Float16 v8h;
typedef __attribute__((ext_vector_type(8)))  float    v8f;

#define T_DIM 4
#define C_DIM 128
#define H_DIM 96
#define W_DIM 96
#define HW    (H_DIM * W_DIM)          // 9216
#define NPIX  (T_DIM * HW)             // 36864
#define NH    4
#define CH    32                       // channels per head
#define WSR   7                        // search window side
#define KNN   16
#define QKV_N 384
#define NCAND 154                      // 7 x 22 candidate band per query row
#define NBAND 160                      // padded to 10 N-tiles

union HF16 { v16h v; v8h h[2]; };

__device__ __forceinline__ v8f wmma_f16(v16h a, v16h b, v8f c) {
  return __builtin_amdgcn_wmma_f32_16x16x32_f16(false, a, false, b,
                                                (short)0, c, false, false);
}

// ---------------------------------------------------------------------------
// K0a: weights for QKV GEMM, transposed to [n=384][k=128] f16
// ---------------------------------------------------------------------------
__global__ __launch_bounds__(256) void prep_wqkv_k(
    const float* __restrict__ wq, const float* __restrict__ wk,
    const float* __restrict__ wv, _Float16* __restrict__ wt) {
  int e = blockIdx.x * 256 + threadIdx.x;          // < 384*128
  int n = e >> 7, c = e & 127;
  float v = (n < 128) ? wq[n * 128 + c]
          : (n < 256) ? wk[(n - 128) * 128 + c]
                      : wv[(n - 256) * 128 + c];
  wt[e] = (_Float16)v;
}

// ---------------------------------------------------------------------------
// K0b: proj3d weights -> Bt[g][ks=144][n=32][cin=32] f16 (ks = k*9+dy*3+dx)
// ---------------------------------------------------------------------------
__global__ __launch_bounds__(256) void prep_proj_k(
    const float* __restrict__ pw, _Float16* __restrict__ bt) {
  int e = blockIdx.x * 256 + threadIdx.x;          // < 4*144*32*32 = 589824
  int cin = e & 31;
  int n   = (e >> 5) & 31;
  int gk  = e >> 10;                               // g*144 + ks
  int g   = gk / 144, ks = gk % 144;
  int k   = ks / 9,  dd = ks % 9;
  int o   = g * 32 + n;
  bt[e] = (_Float16)pw[((size_t)(o * 32 + cin) * 16 + k) * 9 + dd];
}

// ---------------------------------------------------------------------------
// K1: LayerNorm over C at each (t,pixel); one wave per pixel, lane-shuffle
// reduction (wave32). Output pixel-major f16 (NPIX x 128).
// ---------------------------------------------------------------------------
__global__ __launch_bounds__(256) void ln_k(
    const float* __restrict__ vid, const float* __restrict__ lw,
    const float* __restrict__ lb, _Float16* __restrict__ xln) {
  int pix  = blockIdx.x * 8 + (threadIdx.x >> 5);
  int lane = threadIdx.x & 31;
  int t = pix / HW, p = pix % HW;
  const float* base = vid + (size_t)t * C_DIM * HW + p;
  float x[4], s = 0.f, s2 = 0.f;
#pragma unroll
  for (int j = 0; j < 4; ++j) {
    x[j] = base[(size_t)(lane + 32 * j) * HW];
    s += x[j]; s2 += x[j] * x[j];
  }
#pragma unroll
  for (int m = 16; m > 0; m >>= 1) {
    s  += __shfl_xor(s,  m, 32);
    s2 += __shfl_xor(s2, m, 32);
  }
  float mean = s * (1.f / 128.f);
  float var  = s2 * (1.f / 128.f) - mean * mean;
  float inv  = rsqrtf(var + 1e-6f);
#pragma unroll
  for (int j = 0; j < 4; ++j) {
    int ch = lane + 32 * j;
    xln[(size_t)pix * C_DIM + ch] =
        (_Float16)((x[j] - mean) * inv * lw[ch] + lb[ch]);
  }
}

// A fragment: row-major f16 A, 16x32 tile. lane m=L&15, h=L>>4 holds
// k in [16h,16h+16) contiguous (matches B layout)
__device__ __forceinline__ v16h load_a_global(const _Float16* A, int lda,
                                              int mt, int kb, int lane) {
  int m = mt * 16 + (lane & 15);
  int h = lane >> 4;
  const _Float16* p = A + (size_t)m * lda + kb;
  HF16 u;
  u.h[0] = *(const v8h*)(p + 16 * h);
  u.h[1] = *(const v8h*)(p + 16 * h + 8);
  return u.v;
}

// B fragment: Bt stored [n][k] row-major; lane n=L&15, h=L>>4 holds
// k in [16h,16h+16) contiguous  (same k order as A fragment)
__device__ __forceinline__ v16h load_b_global(const _Float16* Bt, int ldb,
                                              int nt, int kb, int lane) {
  int n = nt * 16 + (lane & 15);
  int h = lane >> 4;
  return *(const v16h*)(Bt + (size_t)n * ldb + kb + 16 * h);
}

// ---------------------------------------------------------------------------
// K2: QKV GEMM. Block = 1 M-tile (16 pixels) x 24 N-tiles; 8 waves x 3 tiles.
// ---------------------------------------------------------------------------
__global__ __launch_bounds__(256) void qkv_gemm_k(
    const _Float16* __restrict__ xln, const _Float16* __restrict__ wt,
    const float* __restrict__ bq, const float* __restrict__ bk,
    const float* __restrict__ bv, _Float16* __restrict__ qkv) {
  int mt   = blockIdx.x;                  // 0..2303
  int wave = threadIdx.x >> 5;
  int lane = threadIdx.x & 31;
  v8f zero = {0, 0, 0, 0, 0, 0, 0, 0};
  v8f acc[3] = {zero, zero, zero};
#pragma unroll
  for (int kt = 0; kt < 4; ++kt) {
    v16h a = load_a_global(xln, C_DIM, mt, kt * 32, lane);
#pragma unroll
    for (int j = 0; j < 3; ++j) {
      v16h b = load_b_global(wt, C_DIM, 3 * wave + j, kt * 32, lane);
      acc[j] = wmma_f16(a, b, acc[j]);
    }
  }
  int h = lane >> 4;
#pragma unroll
  for (int j = 0; j < 3; ++j) {
    int n = (3 * wave + j) * 16 + (lane & 15);
    float bias = (n < 128) ? bq[n] : (n < 256) ? bk[n - 128] : bv[n - 256];
#pragma unroll
    for (int r = 0; r < 8; ++r) {
      int m = mt * 16 + 8 * h + r;        // C layout: M = r + 8*(lane>=16)
      qkv[(size_t)m * QKV_N + n] = (_Float16)(acc[j][r] + bias);
    }
  }
}

// ---------------------------------------------------------------------------
// K3: non-local search via unfold-GEMM on WMMA.
// Block = (t, head, 16x16 query tile). LDS (dynamic, 139520 B):
//   qs  : 18x18x32 f16 halo tile of q
//   kls : 24x24x32 f16 halo tile of k
//   sbuf: 8 waves x (16 queries x 160 candidates) f32 score bands
// Each wave processes 2 query rows. Per row: 9 k-steps (patch offsets d)
// x 10 N-tiles of candidates -> 90 WMMA; then lanes 0..15 do sorted top-16
// + softmax over the 49 valid candidates (n = oy*22 + m + ox).
// ---------------------------------------------------------------------------
__global__ __launch_bounds__(256) void nl_search_k(
    const _Float16* __restrict__ qkv, float* __restrict__ wts,
    signed char* __restrict__ ids) {
  extern __shared__ __align__(32) _Float16 smem[];
  _Float16* qs  = smem;                        // 18*18*32 halves
  _Float16* kls = smem + 18 * 18 * CH;         // 24*24*32 halves
  float* sbuf = (float*)(smem + 18 * 18 * CH + 24 * 24 * CH);

  int bi  = blockIdx.x;
  int t   = bi / 144;
  int rem = bi % 144;
  int ty0 = (rem / 24) * 16;
  int tx0 = ((rem / 4) % 6) * 16;
  int hd  = rem & 3;
  int tid = threadIdx.x;

  // stage q (halo r=1) and k (halo r=4), zero outside image
  for (int u = tid; u < 324; u += 256) {
    int i = u / 18, j = u % 18;
    int ay = ty0 - 1 + i, ax = tx0 - 1 + j;
    _Float16* dst = &qs[u * CH];
    if (ay >= 0 && ay < H_DIM && ax >= 0 && ax < W_DIM) {
      const _Float16* src =
          qkv + ((size_t)(t * HW + ay * W_DIM + ax)) * QKV_N + hd * CH;
#pragma unroll
      for (int c = 0; c < 4; ++c) ((v8h*)dst)[c] = ((const v8h*)src)[c];
    } else {
      v8h z = {0, 0, 0, 0, 0, 0, 0, 0};
#pragma unroll
      for (int c = 0; c < 4; ++c) ((v8h*)dst)[c] = z;
    }
  }
  for (int u = tid; u < 576; u += 256) {
    int i = u / 24, j = u % 24;
    int ay = ty0 - 4 + i, ax = tx0 - 4 + j;
    _Float16* dst = &kls[u * CH];
    if (ay >= 0 && ay < H_DIM && ax >= 0 && ax < W_DIM) {
      const _Float16* src =
          qkv + ((size_t)(t * HW + ay * W_DIM + ax)) * QKV_N + 128 + hd * CH;
#pragma unroll
      for (int c = 0; c < 4; ++c) ((v8h*)dst)[c] = ((const v8h*)src)[c];
    } else {
      v8h z = {0, 0, 0, 0, 0, 0, 0, 0};
#pragma unroll
      for (int c = 0; c < 4; ++c) ((v8h*)dst)[c] = z;
    }
  }
  __syncthreads();

  int wave = tid >> 5;
  int lane = tid & 31;
  int h    = lane >> 4;
  int ln15 = lane & 15;
  float* sb = sbuf + wave * 16 * NBAND;
  v8f zero = {0, 0, 0, 0, 0, 0, 0, 0};

  for (int rr = 0; rr < 2; ++rr) {
    int ry = wave + 8 * rr;                 // query row within tile
    v8f acc[10];
#pragma unroll
    for (int j = 0; j < 10; ++j) acc[j] = zero;

    for (int ks = 0; ks < 9; ++ks) {        // patch offset d = (dy,dx)
      int dy = ks / 3, dxx = ks % 3;
      const _Float16* ap = &qs[((ry + dy) * 18 + ln15 + dxx) * CH];
      HF16 u;
      u.h[0] = *(const v8h*)(ap + 16 * h);
      u.h[1] = *(const v8h*)(ap + 16 * h + 8);
#pragma unroll
      for (int j = 0; j < 10; ++j) {
        int cand = j * 16 + ln15;
        if (cand > NCAND - 1) cand = NCAND - 1;   // clamp pad lanes
        int cy = cand / 22, cx = cand % 22;
        v16h b = *(const v16h*)(
            kls + ((size_t)((ry + cy + dy) * 24 + cx + dxx)) * CH + 16 * h);
        acc[j] = wmma_f16(u.v, b, acc[j]);
      }
    }
    // dump score band: C layout lane -> (query m = r+8h, cand n)
#pragma unroll
    for (int j = 0; j < 10; ++j)
#pragma unroll
      for (int r = 0; r < 8; ++r)
        sb[(8 * h + r) * NBAND + j * 16 + ln15] = acc[j][r];
    __syncthreads();

    // top-16 + softmax: lanes 0..15 each own one query of this row
    if (lane < 16) {
      int m = lane;
      float sc[KNN];
      int   id[KNN];
#pragma unroll
      for (int p = 0; p < KNN; ++p) { sc[p] = -1e30f; id[p] = 0; }
      for (int off = 0; off < WSR * WSR; ++off) {
        int oy = off / WSR, ox = off % WSR;
        float sco = sb[m * NBAND + oy * 22 + m + ox];
        if (sco > sc[KNN - 1]) {
          sc[KNN - 1] = sco; id[KNN - 1] = off;
#pragma unroll
          for (int p = KNN - 1; p > 0; --p) {
            float a0 = sc[p - 1], a1 = sc[p];
            int   b0 = id[p - 1], b1 = id[p];
            bool  sw = a1 > a0;
            sc[p - 1] = sw ? a1 : a0; sc[p] = sw ? a0 : a1;
            id[p - 1] = sw ? b1 : b0; id[p] = sw ? b0 : b1;
          }
        }
      }
      float mx = sc[0], sum = 0.f, ev[KNN];
#pragma unroll
      for (int p = 0; p < KNN; ++p) { ev[p] = __expf(sc[p] - mx); sum += ev[p]; }
      float inv = 1.f / sum;
      int ay = ty0 + ry, ax = tx0 + m;
      size_t base = ((size_t)((t * NH + hd) * HW) + ay * W_DIM + ax) * KNN;
#pragma unroll
      for (int p = 0; p < KNN; ++p) {
        wts[base + p] = ev[p] * inv;
        ids[base + p] = (signed char)id[p];
      }
    }
    __syncthreads();
  }
}

// ---------------------------------------------------------------------------
// K4: gather + weight + grouped Conv3d as implicit WMMA GEMM.
// Block = (t, 16x16 tile, group). Reduction K = 16(k)*9(dy,dx)*32(cin)=4608,
// processed in 8 chunks of 2 k-values staged in LDS (18x18 halo x 2 x 32 f16).
// 8 waves x (2 M-tiles x 2 N-tiles) accumulators.
// ---------------------------------------------------------------------------
__global__ __launch_bounds__(256) void proj_gemm_k(
    const _Float16* __restrict__ qkv, const _Float16* __restrict__ bt,
    const float* __restrict__ wts, const signed char* __restrict__ ids,
    const float* __restrict__ pb, float* __restrict__ out) {
  __shared__ __align__(32) _Float16 stage[324 * 2 * CH];   // 41472 B

  int bi  = blockIdx.x;
  int g   = bi & 3;
  int tx0 = ((bi >> 2) % 6) * 16;
  int ty0 = ((bi >> 2) / 6 % 6) * 16;
  int t   = bi / 144;
  int tid  = threadIdx.x;
  int wave = tid >> 5;
  int lane = tid & 31;
  int h    = lane >> 4;

  v8f zero = {0, 0, 0, 0, 0, 0, 0, 0};
  v8f acc[2][2] = {{zero, zero}, {zero, zero}};

  for (int kc = 0; kc < 8; ++kc) {
    // ---- stage weighted gathered v patches for k = {2kc, 2kc+1} ----
    for (int u = tid; u < 648; u += 256) {
      int kk  = u & 1;
      int pos = u >> 1;                       // 0..323
      int i = pos / 18, j = pos % 18;
      int ay = ty0 - 1 + i, ax = tx0 - 1 + j;
      _Float16* dst = &stage[(pos * 2 + kk) * CH];
      bool inside = (ay >= 0 && ay < H_DIM && ax >= 0 && ax < W_DIM);
      if (inside) {
        int kglob = kc * 2 + kk;
        size_t qb = ((size_t)((t * NH + g) * HW) + ay * W_DIM + ax) * KNN + kglob;
        float wgt = wts[qb];
        int ind = (int)ids[qb];
        int vy = ay + ind / WSR - 3;
        int vx = ax + ind % WSR - 3;
        if (vy >= 0 && vy < H_DIM && vx >= 0 && vx < W_DIM) {
          const _Float16* src =
              qkv + ((size_t)(t * HW + vy * W_DIM + vx)) * QKV_N + 256 + g * CH;
          _Float16 wh = (_Float16)wgt;       // packed v_pk_mul_f16 path
#pragma unroll
          for (int c = 0; c < 4; ++c)
            ((v8h*)dst)[c] = ((const v8h*)src)[c] * wh;
          continue;
        }
      }
      v8h z = {0, 0, 0, 0, 0, 0, 0, 0};
#pragma unroll
      for (int c = 0; c < 4; ++c) ((v8h*)dst)[c] = z;
    }
    __syncthreads();

    // ---- 18 WMMA K-steps over this chunk: (kk, dy, dx), 32 cin each ----
    for (int ks2 = 0; ks2 < 18; ++ks2) {
      int ks = kc * 18 + ks2;                 // global 0..143
      int kk = ks2 / 9;
      int dd = ks2 % 9;
      int dy = dd / 3, dxx = dd % 3;
      const _Float16* brow = bt + ((size_t)(g * 144 + ks) * 32) * 32;
      v16h b0 = *(const v16h*)(brow + (size_t)((lane & 15)) * 32 + 16 * h);
      v16h b1 = *(const v16h*)(brow + (size_t)(16 + (lane & 15)) * 32 + 16 * h);
#pragma unroll
      for (int mi = 0; mi < 2; ++mi) {
        int mt = 2 * wave + mi;               // spatial row in tile
        int i = mt + dy;                      // halo coords
        int j = (lane & 15) + dxx;
        const _Float16* ap = &stage[((i * 18 + j) * 2 + kk) * CH];
        HF16 u;
        u.h[0] = *(const v8h*)(ap + 16 * h);
        u.h[1] = *(const v8h*)(ap + 16 * h + 8);
        acc[mi][0] = wmma_f16(u.v, b0, acc[mi][0]);
        acc[mi][1] = wmma_f16(u.v, b1, acc[mi][1]);
      }
    }
    __syncthreads();
  }

  // ---- epilogue: bias + f32 store to (t, ch, y, x) ----
#pragma unroll
  for (int ni = 0; ni < 2; ++ni) {
    int n = g * 32 + ni * 16 + (lane & 15);
    float bias = pb[n];
#pragma unroll
    for (int mi = 0; mi < 2; ++mi) {
      int yy = ty0 + 2 * wave + mi;
#pragma unroll
      for (int r = 0; r < 8; ++r) {
        int xx = tx0 + 8 * h + r;
        out[((size_t)(t * C_DIM + n)) * HW + yy * W_DIM + xx] =
            acc[mi][ni][r] + bias;
      }
    }
  }
}

// ---------------------------------------------------------------------------
// Workspace layout (bytes, 256-aligned)
// ---------------------------------------------------------------------------
static constexpr size_t OFF_XLN = 0;                         //  9437184
static constexpr size_t OFF_QKV = OFF_XLN + (size_t)NPIX * C_DIM * 2;
static constexpr size_t OFF_WT  = OFF_QKV + (size_t)NPIX * QKV_N * 2;
static constexpr size_t OFF_BT  = OFF_WT + (size_t)QKV_N * C_DIM * 2;
static constexpr size_t OFF_WTS = OFF_BT + (size_t)4 * 144 * 32 * 32 * 2;
static constexpr size_t OFF_IDS = OFF_WTS + (size_t)T_DIM * NH * HW * KNN * 4;

// dynamic LDS for nl_search_k: qs + kls (f16) + 8 wave score bands (f32)
static constexpr size_t SEARCH_LDS =
    (size_t)(18 * 18 * CH + 24 * 24 * CH) * 2 + (size_t)8 * 16 * NBAND * 4;

extern "C" void kernel_launch(void* const* d_in, const int* in_sizes, int n_in,
                              void* d_out, int out_size, void* d_ws,
                              size_t ws_size, hipStream_t stream) {
  const float* vid = (const float*)d_in[0];
  const float* lnw = (const float*)d_in[1];
  const float* lnb = (const float*)d_in[2];
  const float* wq  = (const float*)d_in[3];
  const float* bq  = (const float*)d_in[4];
  const float* wk  = (const float*)d_in[5];
  const float* bk  = (const float*)d_in[6];
  const float* wv  = (const float*)d_in[7];
  const float* bv  = (const float*)d_in[8];
  const float* pw  = (const float*)d_in[9];
  const float* pb  = (const float*)d_in[10];
  float* out = (float*)d_out;
  char*  ws  = (char*)d_ws;

  _Float16*    xln  = (_Float16*)(ws + OFF_XLN);
  _Float16*    qkv  = (_Float16*)(ws + OFF_QKV);
  _Float16*    wt   = (_Float16*)(ws + OFF_WT);
  _Float16*    bt   = (_Float16*)(ws + OFF_BT);
  float*       wts  = (float*)(ws + OFF_WTS);
  signed char* ids  = (signed char*)(ws + OFF_IDS);

  prep_wqkv_k<<<(QKV_N * C_DIM) / 256, 256, 0, stream>>>(wq, wk, wv, wt);
  prep_proj_k<<<(4 * 144 * 32 * 32) / 256, 256, 0, stream>>>(pw, bt);
  ln_k<<<NPIX / 8, 256, 0, stream>>>(vid, lnw, lnb, xln);
  qkv_gemm_k<<<NPIX / 16, 256, 0, stream>>>(xln, wt, bq, bk, bv, qkv);
  nl_search_k<<<T_DIM * NH * 36, 256, SEARCH_LDS, stream>>>(qkv, wts, ids);
  proj_gemm_k<<<T_DIM * 36 * NH, 256, 0, stream>>>(qkv, bt, wts, ids, pb, out);
}
